// Qwen3_5MoE_3796751089963
// MI455X (gfx1250) — matrix operand order repaired and checked by reading, not compiled, not run on hardware
//
#include <hip/hip_runtime.h>
#include <hip/hip_bf16.h>
#include <stdint.h>

#define NE 8
#define HD 2048
#define ID 1408
#define NT 1024

#define BM 128
#define BN 64
#define BK 64

#if defined(__has_builtin)
#if __has_builtin(__builtin_amdgcn_global_load_async_to_lds_b128) && \
    __has_builtin(__builtin_amdgcn_s_wait_asynccnt)
#define HAVE_ASYNC_LDS 1
#endif
#endif

typedef __attribute__((ext_vector_type(16))) __bf16 bf16x16;
typedef __attribute__((ext_vector_type(8)))  float  f32x8;

union Frag { bf16x16 v; uint32_t u[8]; };

__device__ inline uint16_t f32_bf16(float f) {
  uint32_t u = __float_as_uint(f);
  u += 0x7FFFu + ((u >> 16) & 1u);   // round-to-nearest-even
  return (uint16_t)(u >> 16);
}
// pack two f32 -> packed bf16 pair: 2 rounding adds + one v_perm_b32
__device__ inline uint32_t pack2(float a, float b) {
  uint32_t ua = __float_as_uint(a);
  uint32_t ub = __float_as_uint(b);
  ua += 0x7FFFu + ((ua >> 16) & 1u);
  ub += 0x7FFFu + ((ub >> 16) & 1u);
  return __builtin_amdgcn_perm(ub, ua, 0x07060302u); // {ub[3],ub[2],ua[3],ua[2]}
}

__device__ inline f32x8 wmma_bf16(bf16x16 a, bf16x16 b, f32x8 c) {
  return __builtin_amdgcn_wmma_f32_16x16x32_bf16(false, a, false, b, (short)0, c,
                                                 false, false);
}

// A-matrix 16x32 bf16 fragment from LDS (row-major).
// lanes 0-15: M=lane, K={0..7,16..23}; lanes 16-31: M=lane-16, K={8..15,24..31}
__device__ inline bf16x16 load_a_frag(const uint16_t* As, int stride, int m0, int k0) {
  int lane = threadIdx.x & 31;
  int hf = lane >> 4, r = lane & 15;
  Frag f;
#pragma unroll
  for (int i = 0; i < 8; ++i) {
    int kk = ((i >> 2) << 4) + (hf << 3) + ((i & 3) << 1);
    f.u[i] = *(const uint32_t*)(As + (m0 + r) * stride + k0 + kk);
  }
  return f.v;
}

// B-matrix 32x16 bf16 fragment; staged column-major (Bs[n][k]).
// lanes 0-15: N=lane, K={0..7,16..23}; lanes 16-31: N=lane-16, K={8..15,24..31}
__device__ inline bf16x16 load_b_frag(const uint16_t* Bs, int stride, int n0, int k0) {
  int lane = threadIdx.x & 31;
  int hf = lane >> 4, n = lane & 15;
  Frag f;
#pragma unroll
  for (int i = 0; i < 8; ++i) {
    int kk = ((i >> 2) << 4) + (hf << 3) + ((i & 3) << 1);
    f.u[i] = *(const uint32_t*)(Bs + (n0 + n) * stride + k0 + kk);
  }
  return f.v;
}

#if HAVE_ASYNC_LDS
// builtin prototype (per hipcc diagnostic): arg0 is int __vector_size__(16) *
// in the global address space; arg1 is the LDS-side pointer; then imm offset,
// imm cpol.
typedef int v4i_ __attribute__((__vector_size__(16)));
typedef __attribute__((address_space(1))) v4i_ gv4i;
typedef __attribute__((address_space(3))) v4i_ lv4i;
__device__ inline void async_b128(const void* g, void* l) {
  __builtin_amdgcn_global_load_async_to_lds_b128((gv4i*)g, (lv4i*)l, 0, 0);
}
#endif

// -------- 1. gating: logits, top-2 (lowest-index ties), renorm weights -------
__global__ __launch_bounds__(256)
void gate_topk_kernel(const float* __restrict__ x, const float* __restrict__ gw,
                      int* __restrict__ tok_exp, float* __restrict__ tok_wt) {
  int wave = threadIdx.x >> 5;
  int lane = threadIdx.x & 31;
  int t = blockIdx.x * 8 + wave;
  float acc[NE];
#pragma unroll
  for (int e = 0; e < NE; ++e) acc[e] = 0.f;
  for (int h = lane; h < HD; h += 32) {
    float xv = x[(size_t)t * HD + h];
#pragma unroll
    for (int e = 0; e < NE; ++e) acc[e] = fmaf(xv, gw[e * HD + h], acc[e]);
  }
#pragma unroll
  for (int e = 0; e < NE; ++e) {
#pragma unroll
    for (int off = 16; off > 0; off >>= 1) acc[e] += __shfl_xor(acc[e], off, 32);
  }
  int m1 = 0;
#pragma unroll
  for (int e = 1; e < NE; ++e) if (acc[e] > acc[m1]) m1 = e;
  int m2 = (m1 == 0) ? 1 : 0;
#pragma unroll
  for (int e = 0; e < NE; ++e) if (e != m1 && acc[e] > acc[m2]) m2 = e;
  float e1 = __expf(acc[m2] - acc[m1]);   // softmax denom cancels in top-k renorm
  float wt0 = 1.f / (1.f + e1);
  float wt1 = e1 / (1.f + e1);
  if (lane == 0) {
    tok_exp[2 * t] = m1; tok_exp[2 * t + 1] = m2;
    tok_wt[2 * t] = wt0; tok_wt[2 * t + 1] = wt1;
  }
}

// -------- 2. deterministic routing: per-expert token lists in token order ----
__global__ void route_kernel(const int* __restrict__ tok_exp,
                             const float* __restrict__ tok_wt,
                             int* __restrict__ slot_tok, int* __restrict__ slot_dst,
                             float* __restrict__ slot_wt,
                             int* __restrict__ counts, int* __restrict__ offs) {
  __shared__ int cnt_s[NE];
  __shared__ int off_s[NE + 1];
  int e = threadIdx.x;
  if (e < NE) {
    int c = 0;
    for (int t = 0; t < NT; ++t)
      if (tok_exp[2 * t] == e || tok_exp[2 * t + 1] == e) ++c;
    cnt_s[e] = c;
  }
  __syncthreads();
  if (e == 0) {
    off_s[0] = 0;
    for (int i = 0; i < NE; ++i) off_s[i + 1] = off_s[i] + cnt_s[i];
  }
  __syncthreads();
  if (e < NE) {
    int s = off_s[e];
    for (int t = 0; t < NT; ++t) {
#pragma unroll
      for (int k = 0; k < 2; ++k) {
        if (tok_exp[2 * t + k] == e) {
          slot_tok[s] = t;
          slot_dst[s] = 2 * t + k;
          slot_wt[s] = tok_wt[2 * t + k];
          ++s;
        }
      }
    }
    counts[e] = cnt_s[e];
    offs[e]   = off_s[e];
  }
}

// -------- 3. grouped GEMM1 + silu: h = silu(x Wgᵀ) * (x Wuᵀ), bf16 out -------
__global__ __launch_bounds__(256)
void gemm1_silu_kernel(const float* __restrict__ x, const float* __restrict__ w1,
                       const int* __restrict__ slot_tok,
                       const int* __restrict__ counts, const int* __restrict__ offs,
                       uint16_t* __restrict__ hbuf) {
  int e = blockIdx.z;
  int cnt = counts[e];
  int mBase = blockIdx.y * BM;
  if (mBase >= cnt) return;
  int off = offs[e];
  int n0 = blockIdx.x * BN;

  __shared__ uint16_t As[BM * BK];   // 16KB
  __shared__ uint16_t Bg[BN * BK];   // 8KB
  __shared__ uint16_t Bu[BN * BK];   // 8KB

  int tid = threadIdx.x;
  int lane = tid & 31, w = tid >> 5;
  int wm = (w >> 1) << 5;   // 0/32/64/96
  int wn = (w & 1) << 5;    // 0/32

  f32x8 accg[2][2], accu[2][2];
#pragma unroll
  for (int ms = 0; ms < 2; ++ms)
#pragma unroll
    for (int ns = 0; ns < 2; ++ns)
#pragma unroll
      for (int i = 0; i < 8; ++i) { accg[ms][ns][i] = 0.f; accu[ms][ns][i] = 0.f; }

  // hoist token gather rows for this thread's 8 A-stage chunks
  const float* arow[8];
#pragma unroll
  for (int j = 0; j < 8; ++j) {
    int c = tid + 256 * j;
    int r = c >> 4;
    int s = mBase + r;
    arow[j] = (s < cnt) ? (x + (size_t)slot_tok[off + s] * HD) : nullptr;
  }

  const size_t w1e = (size_t)e * (2 * ID) * HD;

  for (int k0 = 0; k0 < HD; k0 += BK) {
    // ---- load phase: batch all global loads for this K-tile ----
    float4 aR[8];
#pragma unroll
    for (int j = 0; j < 8; ++j) {
      int c = tid + 256 * j;
      int kk = (c & 15) << 2;
      if (arow[j]) aR[j] = *(const float4*)(arow[j] + k0 + kk);
      else { aR[j].x = aR[j].y = aR[j].z = aR[j].w = 0.f; }
    }
    float4 gR[4], uR[4];
#pragma unroll
    for (int j = 0; j < 4; ++j) {
      int c = tid + 256 * j;
      int n = c >> 4, kk = (c & 15) << 2;
      const float* pg = w1 + w1e + (size_t)(n0 + n) * HD + k0 + kk;
      const float* pu = w1 + w1e + (size_t)(ID + n0 + n) * HD + k0 + kk;
      if (k0 + BK < HD) {
        __builtin_prefetch(pg + BK, 0, 0);
        __builtin_prefetch(pu + BK, 0, 0);
      }
      gR[j] = *(const float4*)pg;
      uR[j] = *(const float4*)pu;
    }
    // ---- convert + store phase ----
#pragma unroll
    for (int j = 0; j < 8; ++j) {
      int c = tid + 256 * j;
      int r = c >> 4, kk = (c & 15) << 2;
      uint32_t* d = (uint32_t*)(As + r * BK + kk);
      d[0] = pack2(aR[j].x, aR[j].y); d[1] = pack2(aR[j].z, aR[j].w);
    }
#pragma unroll
    for (int j = 0; j < 4; ++j) {
      int c = tid + 256 * j;
      int n = c >> 4, kk = (c & 15) << 2;
      uint32_t* dg = (uint32_t*)(Bg + n * BK + kk);
      dg[0] = pack2(gR[j].x, gR[j].y); dg[1] = pack2(gR[j].z, gR[j].w);
      uint32_t* du = (uint32_t*)(Bu + n * BK + kk);
      du[0] = pack2(uR[j].x, uR[j].y); du[1] = pack2(uR[j].z, uR[j].w);
    }
    __syncthreads();
    // ---- WMMA phase ----
#pragma unroll
    for (int kt = 0; kt < BK; kt += 32) {
      bf16x16 a0 = load_a_frag(As, BK, wm, kt);
      bf16x16 a1 = load_a_frag(As, BK, wm + 16, kt);
#pragma unroll
      for (int ns = 0; ns < 2; ++ns) {
        bf16x16 bg = load_b_frag(Bg, BK, wn + (ns << 4), kt);
        accg[0][ns] = wmma_bf16(a0, bg, accg[0][ns]);
        accg[1][ns] = wmma_bf16(a1, bg, accg[1][ns]);
        bf16x16 bu = load_b_frag(Bu, BK, wn + (ns << 4), kt);
        accu[0][ns] = wmma_bf16(a0, bu, accu[0][ns]);
        accu[1][ns] = wmma_bf16(a1, bu, accu[1][ns]);
      }
    }
    __syncthreads();
  }

  int hf = lane >> 4, nn = lane & 15;
#pragma unroll
  for (int ms = 0; ms < 2; ++ms) {
#pragma unroll
    for (int ns = 0; ns < 2; ++ns) {
      int col = n0 + wn + (ns << 4) + nn;          // f index < ID
#pragma unroll
      for (int i = 0; i < 8; ++i) {
        int m = mBase + wm + (ms << 4) + i + (hf << 3);
        if (m < cnt) {
          float g = accg[ms][ns][i], u = accu[ms][ns][i];
          float hv = g / (1.f + __expf(-g)) * u;   // silu(g) * u
          hbuf[(size_t)(off + m) * ID + col] = f32_bf16(hv);
        }
      }
    }
  }
}

// -------- 4. grouped GEMM2: y = (h W2ᵀ) * route_weight -> ybuf[tok*2+k] -----
__global__ __launch_bounds__(256)
void gemm2_kernel(const uint16_t* __restrict__ hbuf, const float* __restrict__ w2,
                  const int* __restrict__ slot_dst, const float* __restrict__ slot_wt,
                  const int* __restrict__ counts, const int* __restrict__ offs,
                  float* __restrict__ ybuf) {
  int e = blockIdx.z;
  int cnt = counts[e];
  int mBase = blockIdx.y * BM;
  if (mBase >= cnt) return;
  int off = offs[e];
  int n0 = blockIdx.x * BN;

  __shared__ uint16_t As[BM * BK];   // 16KB
  __shared__ uint16_t Bs[BN * BK];   // 8KB

  int tid = threadIdx.x;
  int lane = tid & 31, w = tid >> 5;
  int wm = (w >> 1) << 5, wn = (w & 1) << 5;

  f32x8 acc[2][2];
#pragma unroll
  for (int ms = 0; ms < 2; ++ms)
#pragma unroll
    for (int ns = 0; ns < 2; ++ns)
#pragma unroll
      for (int i = 0; i < 8; ++i) acc[ms][ns][i] = 0.f;

  // hoist hbuf rows for this thread's 4 A-stage chunks (128x64 bf16 = 1024x16B)
  const uint16_t* arow[4];
#pragma unroll
  for (int j = 0; j < 4; ++j) {
    int c = tid + 256 * j;
    int r = c >> 3;
    int s = mBase + r;
    arow[j] = (s < cnt) ? (hbuf + (size_t)(off + s) * ID) : nullptr;
  }

  const size_t w2e = (size_t)e * HD * ID;

  for (int k0 = 0; k0 < ID; k0 += BK) {
    // ---- A tile: bf16 rows, direct async DMA to LDS when available ----
#pragma unroll
    for (int j = 0; j < 4; ++j) {
      int c = tid + 256 * j;
      int r = c >> 3, kk = (c & 7) << 3;
      uint16_t* dst = As + r * BK + kk;
      if (arow[j]) {
#if HAVE_ASYNC_LDS
        async_b128(arow[j] + k0 + kk, dst);
#else
        *(uint4*)dst = *(const uint4*)(arow[j] + k0 + kk);
#endif
      } else {
        uint4 z = {0u, 0u, 0u, 0u};
        *(uint4*)dst = z;
      }
    }
    // ---- B tile: w2 64x64 f32 -> bf16 (column-major in LDS) ----
    float4 bR[4];
#pragma unroll
    for (int j = 0; j < 4; ++j) {
      int c = tid + 256 * j;
      int n = c >> 4, kk = (c & 15) << 2;
      const float* pb = w2 + w2e + (size_t)(n0 + n) * ID + k0 + kk;
      if (k0 + BK < ID) __builtin_prefetch(pb + BK, 0, 0);
      bR[j] = *(const float4*)pb;
    }
#pragma unroll
    for (int j = 0; j < 4; ++j) {
      int c = tid + 256 * j;
      int n = c >> 4, kk = (c & 15) << 2;
      uint32_t* d = (uint32_t*)(Bs + n * BK + kk);
      d[0] = pack2(bR[j].x, bR[j].y); d[1] = pack2(bR[j].z, bR[j].w);
    }
#if HAVE_ASYNC_LDS
    __builtin_amdgcn_s_wait_asynccnt(0);
#endif
    __syncthreads();
    // ---- WMMA phase ----
#pragma unroll
    for (int kt = 0; kt < BK; kt += 32) {
      bf16x16 a0 = load_a_frag(As, BK, wm, kt);
      bf16x16 a1 = load_a_frag(As, BK, wm + 16, kt);
#pragma unroll
      for (int ns = 0; ns < 2; ++ns) {
        bf16x16 b = load_b_frag(Bs, BK, wn + (ns << 4), kt);
        acc[0][ns] = wmma_bf16(a0, b, acc[0][ns]);
        acc[1][ns] = wmma_bf16(a1, b, acc[1][ns]);
      }
    }
    __syncthreads();
  }

  int hf = lane >> 4, nn = lane & 15;
#pragma unroll
  for (int ms = 0; ms < 2; ++ms) {
#pragma unroll
    for (int ns = 0; ns < 2; ++ns) {
      int col = n0 + wn + (ns << 4) + nn;
#pragma unroll
      for (int i = 0; i < 8; ++i) {
        int m = mBase + wm + (ms << 4) + i + (hf << 3);
        if (m < cnt) {
          float wgt = slot_wt[off + m];
          int dst = slot_dst[off + m];
          ybuf[(size_t)dst * HD + col] = acc[ms][ns][i] * wgt;
        }
      }
    }
  }
}

// -------- 5. combine: out[t] = y(t, top0) + y(t, top1) (weights applied) ----
__global__ __launch_bounds__(256)
void combine_kernel(const float* __restrict__ ybuf, float* __restrict__ out) {
  int idx = blockIdx.x * 256 + threadIdx.x;
  int t = idx >> 11;        // H = 2048
  int h = idx & (HD - 1);
  out[idx] = ybuf[(size_t)(2 * t) * HD + h] + ybuf[(size_t)(2 * t + 1) * HD + h];
}

extern "C" void kernel_launch(void* const* d_in, const int* in_sizes, int n_in,
                              void* d_out, int out_size, void* d_ws, size_t ws_size,
                              hipStream_t stream) {
  (void)in_sizes; (void)n_in; (void)out_size; (void)ws_size;
  const float* x  = (const float*)d_in[0];
  const float* w1 = (const float*)d_in[1];
  const float* w2 = (const float*)d_in[2];
  const float* gw = (const float*)d_in[3];
  float* out = (float*)d_out;
  char* ws = (char*)d_ws;

  int*      tok_exp  = (int*)(ws + 0);          // 2048 int
  float*    tok_wt   = (float*)(ws + 8192);     // 2048 f32
  int*      slot_tok = (int*)(ws + 16384);      // 2048 int
  int*      slot_dst = (int*)(ws + 24576);      // 2048 int
  float*    slot_wt  = (float*)(ws + 32768);    // 2048 f32
  int*      counts   = (int*)(ws + 40960);      // 8 int
  int*      offs     = (int*)(ws + 41216);      // 8 int
  uint16_t* hbuf     = (uint16_t*)(ws + 49152);               // 2048*1408 bf16
  float*    ybuf     = (float*)(ws + 49152 + (size_t)2048 * ID * 2);  // 2048*2048 f32

  gate_topk_kernel<<<NT / 8, 256, 0, stream>>>(x, gw, tok_exp, tok_wt);
  route_kernel<<<1, 32, 0, stream>>>(tok_exp, tok_wt, slot_tok, slot_dst, slot_wt,
                                     counts, offs);
  gemm1_silu_kernel<<<dim3(ID / BN, NT / BM, NE), 256, 0, stream>>>(
      x, w1, slot_tok, counts, offs, hbuf);
  gemm2_kernel<<<dim3(HD / BN, NT / BM, NE), 256, 0, stream>>>(
      hbuf, w2, slot_dst, slot_wt, counts, offs, ybuf);
  combine_kernel<<<(NT * HD) / 256, 256, 0, stream>>>(ybuf, out);
}
